// NAM_87084756893941
// MI455X (gfx1250) — matrix operand order repaired and checked by reading, not compiled, not run on hardware
//
#include <hip/hip_runtime.h>

#define B_ 2048
#define F_ 128
#define U_ 1024
#define V_ 512
#define C_ 10

typedef _Float16 v16h __attribute__((ext_vector_type(16)));
typedef _Float16 v8h  __attribute__((ext_vector_type(8)));
typedef float    v8f  __attribute__((ext_vector_type(8)));

// ---- tiling parameters ----
#define VN      64            // V columns resident per chunk
#define NVCHUNK (V_ / VN)     // 8
#define BM      128           // batch rows per tile (8 M-tiles)
#define BSLICE  4             // batch slices (blocks per feature)
#define BPS     (B_ / BSLICE) // 512 rows per block
#define NBT     (BPS / BM)    // 4 batch tiles
#define KT      32            // WMMA K per step
#define NKS     (U_ / KT)     // 32 K-steps

// ---- LDS layout (bytes) ----
#define OFF_W1S 0                       // 32 ktiles * 4 ntiles * 1KB = 131072
#define OFF_H2S 131072                  // 128*64 f16                = 16384
#define OFF_EWH (OFF_H2S + 16384)       // 1024 f16                  = 2048
#define OFF_W2S (OFF_EWH + 2048)        // 64*10*4                   = 2560
#define LDS_BYTES (OFF_W2S + 2560)      // 152064 B (< 160KB)

static __device__ __forceinline__ v8h clamp01(v8h v) {
    const v8h z = {};
    const v8h o = z + (_Float16)1.0f;   // splat 1.0
    return __builtin_elementwise_min(__builtin_elementwise_max(v, z), o);
}

// block = (batch-slice, feature), 256 threads = 8 waves.
// Wave w: M-tiles {2*(w&3), 2*(w&3)+1}, N-tiles {2*(w>>2), 2*(w>>2)+1}.
// 8 M-tiles x 4 N-tiles = 32 WMMA tiles = 8 waves x 4 tiles.
__global__ __launch_bounds__(256)
void nam_main(const float* __restrict__ x,     const float* __restrict__ exu_w,
              const float* __restrict__ exu_b, const float* __restrict__ W1,
              const float* __restrict__ b1,    const float* __restrict__ W2,
              float* __restrict__ wlog)
{
    extern __shared__ char smem[];
    _Float16* W1s = (_Float16*)(smem + OFF_W1S);
    _Float16* h2s = (_Float16*)(smem + OFF_H2S);
    _Float16* ewh = (_Float16*)(smem + OFF_EWH);
    float*    w2s = (float*)(smem + OFF_W2S);

    const int tid   = threadIdx.x;
    const int lane  = tid & 31;
    const int wave  = tid >> 5;
    const int f     = blockIdx.y;
    const int bbase = blockIdx.x * BPS;

    const int mt0 = 2 * (wave & 3), mt1 = mt0 + 1;   // M-tiles
    const int np  = wave >> 2;                       // N-tile pair
    const int nt0 = 2 * np, nt1 = 2 * np + 1;
    const int g   = lane >> 4;                       // lane group (kk runs)
    const int mm  = lane & 15;
    const int m0  = mt0 * 16 + mm;                   // A-frag rows for this lane
    const int m1  = mt1 * 16 + mm;

    // Stage f16 exp(exu_w[f,:]) once per block.
    const float xb = exu_b[f];
    for (int u = tid; u < U_; u += 256)
        ewh[u] = (_Float16)__expf(exu_w[f * U_ + u]);

    for (int vc = 0; vc < NVCHUNK; ++vc) {
        const int v0 = vc * VN;
        __syncthreads();  // prior chunk's readers of W1s/w2s done (covers ewh on vc==0)

        // Stage W1[f, :, v0:v0+VN] as f16 in WMMA-B fragment order, vectorized:
        // each thread converts 8 column-strided f32 -> one ds_store_b128.
        // B tile (32K x 16N): lane = nn + 16*((kk>>3)&1), elem = 8*(kk>>4) + (kk&7).
        for (int j = 0; j < 32; ++j) {
            const int id   = tid + 256 * j;       // 0..8191 chunk id
            const int n    = id & 63;
            const int rest = id >> 6;             // 0..127
            const int eh   = rest & 1;            // elem half (0..7 / 8..15)
            const int gg   = (rest >> 1) & 1;     // (kk>>3)&1
            const int kt   = rest >> 2;           // 0..31
            const int u0   = kt * 32 + eh * 16 + gg * 8;
            const float* src = W1 + ((size_t)(f * U_ + u0)) * V_ + v0 + n;
            v8h t;
            #pragma unroll
            for (int e = 0; e < 8; ++e)
                t[e] = (_Float16)src[(size_t)e * V_];
            const int ntile = n >> 4, nn = n & 15;
            const int ln = nn + 16 * gg;
            *(v8h*)(W1s + (size_t)(kt * 4 + ntile) * 512 + ln * 16 + eh * 8) = t;
        }
        // Stage W2[f, v0:v0+VN, :] (row-major [VN][C]).
        for (int i = tid; i < VN * C_; i += 256) {
            const int n = i / C_, c = i - n * C_;
            w2s[i] = W2[((size_t)(f * V_ + v0 + n)) * C_ + c];
        }
        const float b1l0 = b1[f * V_ + v0 + nt0 * 16 + mm];
        const float b1l1 = b1[f * V_ + v0 + nt1 * 16 + mm];
        __syncthreads();

        for (int bt = 0; bt < NBT; ++bt) {
            const int row0 = bbase + bt * BM;
            // Per-lane ExU inputs, shared across the whole K-loop.
            const _Float16 xc0 = (_Float16)(x[(size_t)(row0 + m0) * F_ + f] - xb);
            const _Float16 xc1 = (_Float16)(x[(size_t)(row0 + m1) * F_ + f] - xb);

            v8f acc00 = {}, acc01 = {}, acc10 = {}, acc11 = {};
            // Barrier-free K-loop: A fragments built in-register from f16 exp table.
            #pragma unroll 2
            for (int ks = 0; ks < NKS; ++ks) {
                const _Float16* ew = ewh + ks * KT + g * 8;
                const v8h elo = *(const v8h*)(ew);        // kk = 8g .. 8g+7
                const v8h ehi = *(const v8h*)(ew + 16);   // kk = 16+8g .. +7
                const v16h a0 = __builtin_shufflevector(
                    clamp01(elo * xc0), clamp01(ehi * xc0),
                    0,1,2,3,4,5,6,7,8,9,10,11,12,13,14,15);
                const v16h a1 = __builtin_shufflevector(
                    clamp01(elo * xc1), clamp01(ehi * xc1),
                    0,1,2,3,4,5,6,7,8,9,10,11,12,13,14,15);
                const v16h b0 = *(const v16h*)(W1s + (size_t)(ks * 4 + nt0) * 512 + lane * 16);
                const v16h b1f = *(const v16h*)(W1s + (size_t)(ks * 4 + nt1) * 512 + lane * 16);
                acc00 = __builtin_amdgcn_wmma_f32_16x16x32_f16(
                            false, a0, false, b0, (short)0, acc00, false, false);
                acc01 = __builtin_amdgcn_wmma_f32_16x16x32_f16(
                            false, a0, false, b1f, (short)0, acc01, false, false);
                acc10 = __builtin_amdgcn_wmma_f32_16x16x32_f16(
                            false, a1, false, b0, (short)0, acc10, false, false);
                acc11 = __builtin_amdgcn_wmma_f32_16x16x32_f16(
                            false, a1, false, b1f, (short)0, acc11, false, false);
            }

            // h2 = relu(acc + b1) -> LDS (f16). C/D layout: row = j + 8*(lane>>4).
            {
                const int r0 = mt0 * 16 + 8 * g;
                const int r1 = mt1 * 16 + 8 * g;
                const int c0 = nt0 * 16 + mm;
                const int c1 = nt1 * 16 + mm;
                #pragma unroll
                for (int j = 0; j < 8; ++j) {
                    h2s[(r0 + j) * VN + c0] = (_Float16)fmaxf(acc00[j] + b1l0, 0.0f);
                    h2s[(r0 + j) * VN + c1] = (_Float16)fmaxf(acc01[j] + b1l1, 0.0f);
                    h2s[(r1 + j) * VN + c0] = (_Float16)fmaxf(acc10[j] + b1l0, 0.0f);
                    h2s[(r1 + j) * VN + c1] = (_Float16)fmaxf(acc11[j] + b1l1, 0.0f);
                }
            }
            __syncthreads();

            // GEMM2: [128 x VN] * [VN x C] -> per-feature partial logits
            // (no atomics: block exclusively owns wlog[f, row0:row0+BM, :]).
            for (int o = tid; o < BM * C_; o += 256) {
                const int mr = o / C_;
                const int c  = o - mr * C_;
                float s = 0.0f;
                #pragma unroll 8
                for (int n = 0; n < VN; ++n)
                    s += (float)h2s[mr * VN + n] * w2s[n * C_ + c];
                const size_t oi = ((size_t)f * B_ + row0 + mr) * C_ + c;
                if (vc == 0) wlog[oi] = s;
                else         wlog[oi] += s;
            }
            __syncthreads();  // protect h2s before next tile's writers
        }
    }
}

// Reduce over features, add biases, softmax. One thread per batch row.
__global__ __launch_bounds__(256)
void nam_final(const float* __restrict__ wlog, const float* __restrict__ b2,
               const float* __restrict__ b_out, float* __restrict__ out)
{
    const int b = blockIdx.x * blockDim.x + threadIdx.x;
    if (b >= B_) return;
    float l[C_];
    #pragma unroll
    for (int c = 0; c < C_; ++c) l[c] = b_out[c];
    for (int f = 0; f < F_; ++f) {
        const float* p = wlog + ((size_t)f * B_ + b) * C_;
        const float* q = b2 + f * C_;
        #pragma unroll
        for (int c = 0; c < C_; ++c) l[c] += p[c] + q[c];
    }
    float mx = l[0];
    #pragma unroll
    for (int c = 1; c < C_; ++c) mx = fmaxf(mx, l[c]);
    float e[C_], sum = 0.0f;
    #pragma unroll
    for (int c = 0; c < C_; ++c) { e[c] = __expf(l[c] - mx); sum += e[c]; }
    const float inv = 1.0f / sum;
    #pragma unroll
    for (int c = 0; c < C_; ++c) out[b * C_ + c] = e[c] * inv;
}

extern "C" void kernel_launch(void* const* d_in, const int* in_sizes, int n_in,
                              void* d_out, int out_size, void* d_ws, size_t ws_size,
                              hipStream_t stream)
{
    (void)in_sizes; (void)n_in; (void)out_size; (void)ws_size;
    const float* x     = (const float*)d_in[0];
    const float* exu_w = (const float*)d_in[1];
    const float* exu_b = (const float*)d_in[2];
    const float* W1    = (const float*)d_in[3];
    const float* b1    = (const float*)d_in[4];
    const float* W2    = (const float*)d_in[5];
    const float* b2    = (const float*)d_in[6];
    const float* b_out = (const float*)d_in[7];
    float* out  = (float*)d_out;
    float* wlog = (float*)d_ws;   // F_*B_*C_ floats = 10.5 MB scratch

    hipFuncSetAttribute(reinterpret_cast<const void*>(nam_main),
                        hipFuncAttributeMaxDynamicSharedMemorySize, LDS_BYTES);

    dim3 grid(BSLICE, F_);
    nam_main<<<grid, 256, LDS_BYTES, stream>>>(x, exu_w, exu_b, W1, b1, W2, wlog);
    nam_final<<<(B_ + 255) / 256, 256, 0, stream>>>(wlog, b2, b_out, out);
}
